// LowRankRotatedSpaceIntervention_4509715660999
// MI455X (gfx1250) — matrix operand order repaired and checked by reading, not compiled, not run on hardware
//
#include <hip/hip_runtime.h>
#include <hip/hip_bf16.h>

// ---------------------------------------------------------------------------
// LowRankRotatedSpaceIntervention, fused for MI455X (gfx1250, wave32, WMMA).
//
//   sel   = subspaces[0]                (128 indices into 512 cols of W)
//   W_sel = W[:, sel]                   (4096 x 128)
//   out   = base + ((source-base) @ W_sel) @ W_sel^T
//
// Memory-bound (~540MB HBM traffic vs ~17 GFLOP). bf16 WMMA, f32 accumulate.
// B operands are staged once per workgroup into LDS via the CDNA5 async-copy
// path (global_load_async_to_lds_b128 + s_wait_asynccnt), double-buffered.
// ---------------------------------------------------------------------------

#define EMBED   4096
#define LOWRANK 512
#define BATCH   8192
#define KSEL    128
#define MTILE   128   // rows per workgroup (8 waves x 16 rows)

typedef __attribute__((ext_vector_type(16))) __bf16 v16bf;
typedef __attribute__((ext_vector_type(8)))  __bf16 v8bf;
typedef __attribute__((ext_vector_type(8)))  float  v8f;
typedef __attribute__((ext_vector_type(4)))  float  v4f;

static __device__ __forceinline__ __bf16 f2bf(float x) { return (__bf16)x; }

// ---------------------------------------------------------------------------
// Kernel 1: gather W[:,sel] -> bf16, pre-swizzled into WMMA B-matrix layout.
//
// B-matrix 32x16 (KxN) bf16 block layout (wave32):
//   lane      = n + 16*((k/8)%2)         (n = 0..15, k = 0..31)
//   halfword  = (k%8) + 8*(k/16)         (16 bf16 = 32 bytes per lane)
// Block storage: block[lane*16 + halfword], 512 bf16 per block.
//
// pB1: GEMM1 B = W_sel      blocks [kt=0..127][nt=0..7]    (K=embed, N=sel)
// pB2: GEMM2 B = W_sel^T    blocks [kt=0..3 ][nt=0..255]   (K=sel,  N=embed)
// ---------------------------------------------------------------------------
__global__ __launch_bounds__(256) void pack_wsel_kernel(
    const float* __restrict__ W, const int* __restrict__ sel,
    __bf16* __restrict__ pB1, __bf16* __restrict__ pB2)
{
    int idx = blockIdx.x * 256 + threadIdx.x;          // 0 .. 2*524288-1
    const int NB1 = EMBED * KSEL;                      // 524288 elements each
    bool second = idx >= NB1;
    int e = second ? idx - NB1 : idx;

    int p    = e & 15;           // halfword within lane
    int lane = (e >> 4) & 31;
    int blk  = e >> 9;
    int n16  = lane & 15;
    int k    = ((p & 8) << 1) | ((lane >> 4) << 3) | (p & 7);   // 0..31 within block

    if (!second) {
        int nt = blk & 7;        // 8 N-tiles over KSEL=128
        int kt = blk >> 3;       // 128 K-tiles over EMBED=4096
        int krow = kt * 32 + k;              // embed index (row of W)
        int ncol = nt * 16 + n16;            // position in sel
        pB1[e] = f2bf(W[(size_t)krow * LOWRANK + sel[ncol]]);
    } else {
        int nt = blk & 255;      // 256 N-tiles over EMBED=4096
        int kt = blk >> 8;       // 4 K-tiles over KSEL=128
        int k2 = kt * 32 + k;                // position in sel
        int n  = nt * 16 + n16;              // embed index (row of W)
        pB2[e] = f2bf(W[(size_t)n * LOWRANK + sel[k2]]);
    }
}

// ---------------------------------------------------------------------------
// Async stage: NBYTES from global -> LDS, whole workgroup (256 threads),
// 16B per lane per op. Tracked by ASYNCcnt; caller waits + barriers.
// NBYTES must be a multiple of 4096 (= 256 threads * 16B).
// ---------------------------------------------------------------------------
template <int NBYTES>
static __device__ __forceinline__ void stage_async(
    const __bf16* __restrict__ gsrc, __bf16* ldst, int tid)
{
    // low 32 bits of a generic pointer to __shared__ == LDS byte address
    unsigned lbase = (unsigned)(size_t)ldst + (unsigned)tid * 16u;
    unsigned long long g = (unsigned long long)(size_t)gsrc;
#pragma unroll
    for (int off = 0; off < NBYTES; off += 256 * 16) {
        asm volatile("global_load_async_to_lds_b128 %0, %1, %2"
                     :: "v"(lbase + (unsigned)off),
                        "v"(off + tid * 16),
                        "s"(g)
                     : "memory");
    }
}

#define WAIT_ASYNC_LE_8() asm volatile("s_wait_asynccnt 0x8" ::: "memory")
#define WAIT_ASYNC_0()    asm volatile("s_wait_asynccnt 0x0" ::: "memory")

// ---------------------------------------------------------------------------
// Kernel 2: fused   out = base + ((source-base) @ W_sel) @ W_sel^T
// One workgroup = 256 threads = 8 waves; wave w owns rows m0 = blk*128+16w.
// ---------------------------------------------------------------------------
__global__ __launch_bounds__(256) void fused_intervention_kernel(
    const float* __restrict__ base, const float* __restrict__ src,
    const __bf16* __restrict__ pB1, const __bf16* __restrict__ pB2,
    float* __restrict__ out)
{
    // Ping-pong staging for B blocks (2 x 32KB) + T strip (128x128 bf16,
    // row stride 136 = 17*8 keeps 16B alignment and rotates LDS banks).
    __shared__ __bf16 Bstage[2][16384];
    __shared__ __bf16 Tlds[MTILE][136];

    const int tid  = threadIdx.x;
    const int lane = tid & 31;
    const int w    = tid >> 5;                  // wave id 0..7
    const int r0   = blockIdx.x * MTILE;        // first row of this block
    const int m0   = r0 + w * 16;               // first row of this wave

    // A-fragment addressing (16-bit A 16x32 layout):
    //   lane<16 : row=lane,    K = {0..7, 16..23}
    //   lane>=16: row=lane-16, K = {8..15, 24..31}
    const int rowA   = m0 + (lane & 15);
    const int kshift = (lane >> 4) * 8;         // 0 or 8
    const float* baseRow = base + (size_t)rowA * EMBED;
    const float* srcRow  = src  + (size_t)rowA * EMBED;

    // ---------------- GEMM1: T = (src-base) @ W_sel  (16 x 128 per wave) ----
    v8f acc[8];
#pragma unroll
    for (int i = 0; i < 8; ++i) acc[i] = v8f{0.f,0.f,0.f,0.f,0.f,0.f,0.f,0.f};

    // 32 chunks of 4 K-tiles; chunk = 4*8 blocks * 512 bf16 = 32KB
    stage_async<32768>(pB1, Bstage[0], tid);
    for (int c = 0; c < 32; ++c) {
        if (c + 1 < 32) {
            stage_async<32768>(pB1 + (size_t)(c + 1) * 16384,
                               Bstage[(c + 1) & 1], tid);
            WAIT_ASYNC_LE_8();   // own chunk-c ops done; c+1 still in flight
        } else {
            WAIT_ASYNC_0();
        }
        __syncthreads();         // chunk c visible from all waves

        const __bf16* buf = Bstage[c & 1];
#pragma unroll
        for (int ki = 0; ki < 4; ++ki) {
            const int k0 = (c * 4 + ki) * 32 + kshift;
            v4f b0 = *(const v4f*)(baseRow + k0);
            v4f b1 = *(const v4f*)(baseRow + k0 + 4);
            v4f b2 = *(const v4f*)(baseRow + k0 + 16);
            v4f b3 = *(const v4f*)(baseRow + k0 + 20);
            v4f s0 = *(const v4f*)(srcRow  + k0);
            v4f s1 = *(const v4f*)(srcRow  + k0 + 4);
            v4f s2 = *(const v4f*)(srcRow  + k0 + 16);
            v4f s3 = *(const v4f*)(srcRow  + k0 + 20);

            v16bf a;
#pragma unroll
            for (int i = 0; i < 4; ++i) {
                a[i]      = f2bf(s0[i] - b0[i]);
                a[4 + i]  = f2bf(s1[i] - b1[i]);
                a[8 + i]  = f2bf(s2[i] - b2[i]);
                a[12 + i] = f2bf(s3[i] - b3[i]);
            }

            const __bf16* bp = buf + ki * 4096 + lane * 16;
#pragma unroll
            for (int nt = 0; nt < 8; ++nt) {
                v16bf b = *(const v16bf*)(bp + nt * 512);
                acc[nt] = __builtin_amdgcn_wmma_f32_16x16x32_bf16(
                    false, a, false, b, (short)0, acc[nt], false, false);
            }
        }
        __syncthreads();         // all waves done reading before overwrite
    }

    // ---------------- T -> LDS (bf16), C/D 16x16 f32 layout ---------------
    {
        const int col16 = lane & 15;
        const int rbase = w * 16 + 8 * (lane >> 4);
#pragma unroll
        for (int nt = 0; nt < 8; ++nt) {
            const int col = nt * 16 + col16;
#pragma unroll
            for (int v = 0; v < 8; ++v)
                Tlds[rbase + v][col] = f2bf(acc[nt][v]);
        }
    }
    __syncthreads();

    // ---------------- Re-read T as A2 fragments (16 x 128 per wave) --------
    v16bf a2[4];
    {
        const int r = w * 16 + (lane & 15);
#pragma unroll
        for (int kk = 0; kk < 4; ++kk) {
            const int c0 = kk * 32 + kshift;
            v8bf lo = *(const v8bf*)&Tlds[r][c0];
            v8bf hi = *(const v8bf*)&Tlds[r][c0 + 16];
            v16bf t;
#pragma unroll
            for (int i = 0; i < 8; ++i) { t[i] = lo[i]; t[8 + i] = hi[i]; }
            a2[kk] = t;
        }
    }

    // ---------------- GEMM2 + epilogue: out = base + T @ W_sel^T -----------
    // 32 column chunks of 8 N-tiles (128 cols); chunk = 4 K-slabs x 8KB = 32KB
    const int col16 = lane & 15;
    const int rhalf = 8 * (lane >> 4);

#pragma unroll
    for (int kt2 = 0; kt2 < 4; ++kt2)
        stage_async<8192>(pB2 + (size_t)(kt2 * 256) * 512,
                          Bstage[0] + kt2 * 4096, tid);

    for (int c = 0; c < 32; ++c) {
        if (c + 1 < 32) {
#pragma unroll
            for (int kt2 = 0; kt2 < 4; ++kt2)
                stage_async<8192>(pB2 + (size_t)(kt2 * 256 + (c + 1) * 8) * 512,
                                  Bstage[(c + 1) & 1] + kt2 * 4096, tid);
            WAIT_ASYNC_LE_8();
        } else {
            WAIT_ASYNC_0();
        }
        __syncthreads();

        const __bf16* buf = Bstage[c & 1];
#pragma unroll
        for (int t = 0; t < 8; ++t) {
            v8f cc = v8f{0.f,0.f,0.f,0.f,0.f,0.f,0.f,0.f};
#pragma unroll
            for (int kt2 = 0; kt2 < 4; ++kt2) {
                v16bf b = *(const v16bf*)(buf + (kt2 * 8 + t) * 512 + lane * 16);
                cc = __builtin_amdgcn_wmma_f32_16x16x32_bf16(
                    false, a2[kt2], false, b, (short)0, cc, false, false);
            }

            const int col  = (c * 8 + t) * 16 + col16;
            const int row0 = m0 + rhalf;
#pragma unroll
            for (int v = 0; v < 8; ++v) {
                const size_t o = (size_t)(row0 + v) * EMBED + col;
                out[o] = base[o] + cc[v];
            }
        }
        __syncthreads();
    }
}

// ---------------------------------------------------------------------------
// Host entry. Inputs (setup_inputs order): base f32, source f32,
// subspaces int, W f32. Workspace: pB1 (1 MiB) + pB2 (1 MiB) bf16.
// ---------------------------------------------------------------------------
extern "C" void kernel_launch(void* const* d_in, const int* in_sizes, int n_in,
                              void* d_out, int out_size, void* d_ws, size_t ws_size,
                              hipStream_t stream) {
    (void)in_sizes; (void)n_in; (void)out_size; (void)ws_size;
    const float* base   = (const float*)d_in[0];
    const float* source = (const float*)d_in[1];
    const int*   subs   = (const int*)d_in[2];   // only row 0 (128 indices) used
    const float* W      = (const float*)d_in[3];
    float*       out    = (float*)d_out;

    __bf16* pB1 = (__bf16*)d_ws;                 // EMBED*KSEL bf16
    __bf16* pB2 = pB1 + (size_t)EMBED * KSEL;    // KSEL*EMBED bf16

    // 2 * 524288 elements, one thread each
    pack_wsel_kernel<<<dim3(2 * EMBED * KSEL / 256), dim3(256), 0, stream>>>(
        W, subs, pB1, pB2);

    fused_intervention_kernel<<<dim3(BATCH / MTILE), dim3(256), 0, stream>>>(
        base, source, pB1, pB2, out);
}
